// Contraction_79508434583947
// MI455X (gfx1250) — matrix operand order repaired and checked by reading, not compiled, not run on hardware
//
#include <hip/hip_runtime.h>

typedef __attribute__((ext_vector_type(16))) _Float16 v16h;
typedef __attribute__((ext_vector_type(8)))  float    v8f;
typedef __attribute__((ext_vector_type(4)))  float    v4f;

// ---------------------------------------------------------------------------
// Problem constants (from reference): N=2048 nodes, C=128 channels, D=9,
// chunks (mul, ir) = (7,1), (11,3), (12,5); out irrep dim 1+3+5 = 9.
// GEMM per chunk: OUT[(n,c),(a,b,i)] = A[(n,c),(k*9+j)] * B[(k*9+j),(a,b,i)]
//   A[c][kk] = x[n,c,j] * W[s,k,c] * mul^-0.25       (kk = k*9 + j)
//   B[kk][m] = u[a,b,j,k,i]                          (m  = (a*9+b)*ir + i)
// K padded to 64 / 128 / 128 (2/4/4 wmma k-steps), M tiled by 16 (6/16/26).
// ---------------------------------------------------------------------------

#define NNODES   2048
#define NCHAN    128
#define OUTROW   729            // 81 * 9  floats per (n,c)
#define FRAG_BYTES 1024         // 32 lanes * 32 B per WMMA-B fragment

// fragment table: chunk0: 6 tiles * 2 ksteps = 12  (base 0)
//                 chunk1: 16 tiles * 4 ksteps = 64 (base 12)
//                 chunk2: 26 tiles * 4 ksteps = 104 (base 76) -> 180 total

// ---------------------------------------------------------------------------
// Prep: build f16 WMMA-B fragments from u (f32) into d_ws.
// B layout (dense 16x16x32, this code's contiguous k order, same as A):
//   lane l, half-slot t:  column N = l%16,  K = (l<16 ? t : 16+t)
// ---------------------------------------------------------------------------
template<int MUL, int IR, int KSTEPS>
__device__ inline void build_frag(const float* __restrict__ u, char* ws,
                                  int f, int base, int lane) {
    const int rel = f - base;
    const int mt  = rel / KSTEPS;
    const int ks  = rel % KSTEPS;
    const int col = mt * 16 + (lane & 15);
    v16h bf;
#pragma unroll
    for (int t = 0; t < 16; ++t) {
        const int K = ks * 32 + ((lane < 16) ? t : t + 16);
        float v = 0.0f;
        if (K < 9 * MUL && col < 81 * IR) {
            const int j   = K % 9;
            const int k   = K / 9;
            const int a   = col / (9 * IR);
            const int rem = col % (9 * IR);
            const int b   = rem / IR;
            const int i   = rem % IR;
            v = u[(((a * 9 + b) * 9 + j) * MUL + k) * IR + i];
        }
        bf[t] = (_Float16)v;
    }
    *(v16h*)(ws + (size_t)f * FRAG_BYTES + lane * 32) = bf;
}

__global__ __launch_bounds__(32) void mace_sc_prep(const float* __restrict__ u0,
                                                   const float* __restrict__ u1,
                                                   const float* __restrict__ u2,
                                                   char* __restrict__ ws) {
    const int f    = blockIdx.x;     // 0..179
    const int lane = threadIdx.x;    // wave32
    if (f < 12)      build_frag<7, 1, 2>(u0, ws, f, 0,  lane);
    else if (f < 76) build_frag<11, 3, 4>(u1, ws, f, 12, lane);
    else             build_frag<12, 5, 4>(u2, ws, f, 76, lane);
}

// ---------------------------------------------------------------------------
// Main: one wave per (node, 16-channel) row-tile.
// A layout (dense 16-bit A 16x32, contiguous k order, same as B):
//   lane l, half-slot t: row M = l%16,
//   K = (l<16 ? t : 16+t) within a 32-K step.
// ---------------------------------------------------------------------------
template<int MUL, int IR, int IOFF, int KSTEPS, int TILES, int WOFFS, int FRAGBASE>
__device__ inline void run_chunk(int lane,
                                 const float* __restrict__ xsh,
                                 const float* __restrict__ wsh,
                                 const char*  __restrict__ ws,
                                 float* __restrict__ cbuf) {
    constexpr int KTOT = 9 * MUL;
    constexpr int MCH  = 81 * IR;
    const int cl = lane & 15;

    // Build A fragments for all k-steps of this chunk.
    v16h af[KSTEPS];
#pragma unroll
    for (int ks = 0; ks < KSTEPS; ++ks) {
        v16h a;
#pragma unroll
        for (int t = 0; t < 16; ++t) {
            const int kk = ks * 32 + ((lane < 16) ? t : t + 16);
            float v = 0.0f;
            if (kk < KTOT) {
                const int j = kk % 9;
                const int k = kk / 9;
                v = xsh[j * 16 + cl] * wsh[WOFFS + k * 16 + cl];
            }
            a[t] = (_Float16)v;
        }
        af[ks] = a;
    }

    // Column tiles: load B fragments (L2-hot), WMMA-accumulate in f32,
    // scatter the 16x16 result tile into the LDS output row buffer.
    for (int mt = 0; mt < TILES; ++mt) {
        v16h bf[KSTEPS];
#pragma unroll
        for (int ks = 0; ks < KSTEPS; ++ks)
            bf[ks] = *(const v16h*)(ws +
                       (size_t)(FRAGBASE + mt * KSTEPS + ks) * FRAG_BYTES +
                       lane * 32);
        v8f acc = {};
#pragma unroll
        for (int ks = 0; ks < KSTEPS; ++ks)
            acc = __builtin_amdgcn_wmma_f32_16x16x32_f16(
                false, af[ks], false, bf[ks], (short)0, acc, false, false);

        const int mcol = mt * 16 + cl;          // C/D layout: column N = lane%16
        if (mcol < MCH) {
            const int q    = mcol / IR;
            const int i    = mcol % IR;
            const int moff = q * 9 + IOFF + i;  // position inside 729-wide row
            const int row0 = (lane < 16) ? 0 : 8;  // VGPR r: M = r (+8 hi half)
#pragma unroll
            for (int r = 0; r < 8; ++r)
                cbuf[(row0 + r) * OUTROW + moff] = acc[r];
        }
    }
}

__global__ __launch_bounds__(32) void mace_sc_main(const float* __restrict__ x,
                                                   const int*   __restrict__ idx,
                                                   const float* __restrict__ w0,
                                                   const float* __restrict__ w1,
                                                   const float* __restrict__ w2,
                                                   const char*  __restrict__ ws,
                                                   float* __restrict__ out) {
    __shared__ float xsh[9 * 16];        // xsh[j*16 + c]
    __shared__ float wsh[(7 + 11 + 12) * 16]; // per-chunk, k*16 + c, pre-scaled
    __shared__ alignas(16) float cbuf[16 * OUTROW]; // 16 ch x 729 (46.6 KB)

    const int lane = threadIdx.x;        // wave32, one wave per block
    const int rt   = blockIdx.x;         // row-tile id, 0..16383
    const int n    = rt >> 3;            // node
    const int c0   = (rt & 7) * 16;      // channel block
    const int s    = idx[n];             // species

    // Stage x[n, c0..c0+15, :] and scaled W[s, :, c0..c0+15] in LDS.
    for (int t = lane; t < 9 * 16; t += 32) {
        const int c = t & 15, j = t >> 4;
        xsh[t] = x[(size_t)(n * NCHAN + c0 + c) * 9 + j];
    }
    for (int t = lane; t < 7 * 16; t += 32) {
        const int c = t & 15, k = t >> 4;
        wsh[t] = w0[(s * 7 + k) * NCHAN + c0 + c] * 0.6147882f;   // 7^-0.25
    }
    for (int t = lane; t < 11 * 16; t += 32) {
        const int c = t & 15, k = t >> 4;
        wsh[112 + t] = w1[(s * 11 + k) * NCHAN + c0 + c] * 0.5491004f; // 11^-0.25
    }
    for (int t = lane; t < 12 * 16; t += 32) {
        const int c = t & 15, k = t >> 4;
        wsh[288 + t] = w2[(s * 12 + k) * NCHAN + c0 + c] * 0.5372849f; // 12^-0.25
    }
    __syncthreads();

    //            MUL IR IOFF KS TILES WOFF FRAGBASE
    run_chunk<     7, 1, 0,   2,  6,    0,   0>(lane, xsh, wsh, ws, cbuf);
    run_chunk<    11, 3, 1,   4, 16,  112,  12>(lane, xsh, wsh, ws, cbuf);
    run_chunk<    12, 5, 4,   4, 26,  288,  76>(lane, xsh, wsh, ws, cbuf);

    __syncthreads();

    // The three chunks exactly tile all 729 slots per (n,c); rows c0..c0+15
    // are contiguous in the output. Block base = (n*128+c0)*729*4 bytes, a
    // multiple of 46656 -> 16B aligned; 11664 floats = 2916 exact float4s.
    // Stream out with non-temporal b128 stores (write-once 764 MB stream;
    // keep L2 for the B-fragment table / u / x working set).
    const size_t gbase = (size_t)(n * NCHAN + c0) * OUTROW;
    const v4f* cb4 = (const v4f*)cbuf;
    v4f*       ob4 = (v4f*)(out + gbase);
    for (int t = lane; t < (16 * OUTROW) / 4; t += 32)
        __builtin_nontemporal_store(cb4[t], ob4 + t);
}

// ---------------------------------------------------------------------------
extern "C" void kernel_launch(void* const* d_in, const int* in_sizes, int n_in,
                              void* d_out, int out_size, void* d_ws, size_t ws_size,
                              hipStream_t stream) {
    (void)in_sizes; (void)n_in; (void)out_size; (void)ws_size;
    // setup_inputs order: node_feats, index, u0, w0, u1, w1, u2, w2
    const float* x  = (const float*)d_in[0];
    const int*   ix = (const int*)  d_in[1];
    const float* u0 = (const float*)d_in[2];
    const float* w0 = (const float*)d_in[3];
    const float* u1 = (const float*)d_in[4];
    const float* w1 = (const float*)d_in[5];
    const float* u2 = (const float*)d_in[6];
    const float* w2 = (const float*)d_in[7];
    float* out = (float*)d_out;
    char*  ws  = (char*)d_ws;   // needs 180 * 1024 = 184320 bytes

    mace_sc_prep<<<180, 32, 0, stream>>>(u0, u1, u2, ws);
    mace_sc_main<<<NNODES * (NCHAN / 16), 32, 0, stream>>>(x, ix, w0, w1, w2, ws, out);
}
